// VelDecoder_52604759441900
// MI455X (gfx1250) — compile-verified, matrix operand order checked by reading, not run on hardware
//
#include <hip/hip_runtime.h>
#include <hip/hip_bf16.h>
#include <cstdint>
#include <cstddef>

typedef _Float16 v16h __attribute__((ext_vector_type(16)));
typedef _Float16 h8   __attribute__((ext_vector_type(8)));
typedef float    v8f  __attribute__((ext_vector_type(8)));

// ---------------- packed weight layout in d_ws (units: halves) ----------------
// Each 32x16 (KxN) f16 B-fragment is 512 halves, laid out as frag[lane*16 + i]
// with lane<16 -> N-col = lane,  halves 0..7 -> K 0..7,  8..15 -> K 16..23
//      lane>=16 -> N-col = lane-16, halves 0..7 -> K 8..15, 8..15 -> K 24..31
// Fragment order within a matrix: frag = nt*KT + kt.
constexpr size_t FR = 512;
constexpr size_t OFF_UP_WIH = 0;                                  // 768x34  KT=2 NT=48
constexpr size_t OFF_UP_WHH = OFF_UP_WIH + (size_t)2 * 48 * FR;   // 768x256 KT=8 NT=48
constexpr size_t OFF_LO_WIH = OFF_UP_WHH + (size_t)8 * 48 * FR;   // 768x46  KT=2 NT=48
constexpr size_t OFF_LO_WHH = OFF_LO_WIH + (size_t)2 * 48 * FR;   // 768x256
constexpr size_t OFF_W3A    = OFF_LO_WHH + (size_t)8 * 48 * FR;   // 256x256 KT=8 NT=16
constexpr size_t OFF_W3L    = OFF_W3A    + (size_t)8 * 16 * FR;
constexpr size_t OFF_W2RA   = OFF_W3L    + (size_t)8 * 16 * FR;   // 64x128  KT=4 NT=4
constexpr size_t OFF_W2LA   = OFF_W2RA   + (size_t)4 * 4 * FR;
constexpr size_t OFF_W2RL   = OFF_W2LA   + (size_t)4 * 4 * FR;
constexpr size_t OFF_W2LL   = OFF_W2RL   + (size_t)4 * 4 * FR;
constexpr size_t OFF_W1RA   = OFF_W2LL   + (size_t)4 * 4 * FR;    // 9x32  KT=1 NT=1
constexpr size_t OFF_W1LA   = OFF_W1RA   + FR;
constexpr size_t OFF_W1RL   = OFF_W1LA   + FR;
constexpr size_t OFF_W1LL   = OFF_W1RL   + FR;
constexpr size_t OFF_W1T    = OFF_W1LL   + FR;                    // 16x32

// ---------------- weight packing: f32 (N,K) row-major -> f16 fragments -------
__global__ void pack_w_kernel(const float* __restrict__ W, _Float16* __restrict__ out,
                              int N, int K, int NT, int KT) {
  int idx = blockIdx.x * blockDim.x + threadIdx.x;
  int total = NT * KT * 512;
  if (idx >= total) return;
  int frag   = idx >> 9;
  int within = idx & 511;
  int lane   = within >> 4;
  int i      = within & 15;
  int nt = frag / KT, kt = frag % KT;
  int n = nt * 16 + (lane & 15);
  int k_in = ((i < 8) ? i : i + 8) + ((lane < 16) ? 0 : 8);
  int k = kt * 32 + k_in;
  float v = (n < N && k < K) ? W[(size_t)n * K + k] : 0.f;
  out[idx] = (_Float16)v;
}

// ---------------- WMMA helpers ----------------
__device__ __forceinline__ v16h wmma_load_a(const _Float16* base, int stride, int kt, int lane) {
  // 16-bit A-matrix 16x32 layout: lane&15 = row M; lane>=16 uses K+8 group.
  int row = lane & 15;
  int khi = (lane & 16) ? 8 : 0;
  const _Float16* p = base + row * stride + kt * 32 + khi;
  h8 lo = *(const h8*)(p);
  h8 hi = *(const h8*)(p + 16);
  v16h a;
#pragma unroll
  for (int i = 0; i < 8; ++i) { a[i] = lo[i]; a[8 + i] = hi[i]; }
  return a;
}
__device__ __forceinline__ v16h wmma_load_b(const _Float16* frag, int lane) {
  return *(const v16h*)(frag + lane * 16);
}
__device__ __forceinline__ v8f bias_init(float bv) {
  v8f c;
#pragma unroll
  for (int i = 0; i < 8; ++i) c[i] = bv;
  return c;
}
#define WMMA_F16(a, b, c) \
  __builtin_amdgcn_wmma_f32_16x16x32_f16(false, (a), false, (b), (short)0, (c), false, false)

__device__ __forceinline__ float sigmoidf_(float x) { return 1.f / (1.f + __expf(-x)); }

struct KParams {
  const float* hu0; const float* hl0; const float* gt; const int* tsteps;
  const _Float16* w;
  const float* bih_u; const float* bhh_u; const float* bih_l; const float* bhh_l;
  const float* b3a; const float* b3l;
  const float* b2[4];   // ra, la, rl, ll
  const float* b1[5];   // ra, la, rl, ll, t
  float* out;
};

// ---------------- persistent per-batch-tile recurrence kernel ----------------
__global__ __launch_bounds__(256, 1) void veldec_main(KParams p) {
  const int tid   = threadIdx.x;
  const int lane  = tid & 31;
  const int wv    = tid >> 5;          // 8 waves per block
  const int rbase = blockIdx.x * 16;   // 16 batch rows per block
  const int T     = p.tsteps[0];

  // LDS (~72 KB; WGP budget is 320 KB)
  __shared__ __align__(16) _Float16 sH[2][16][256];  // GRU hidden states (f16 A-operand)
  __shared__ __align__(16) float    sX[16][64];      // running x (residual stream)
  __shared__ __align__(16) _Float16 sXU[16][64];     // padded upper GRU input
  __shared__ __align__(16) _Float16 sXL[16][64];     // padded lower GRU input
  __shared__ __align__(16) float    sA[16][512];     // f32 scratch (gin|ghn / ub|lb / l2 / l1)
  __shared__ __align__(16) _Float16 sB[16][512];     // f16 scratch (g_rz / mid16 / l2f16)

  // ---- init state ----
  for (int e = tid; e < 16 * 64; e += 256) {
    int m = e >> 6, c = e & 63;
    sX[m][c] = p.gt[(size_t)(rbase + m) * T * 64 + c];   // gt[:,0,:]
  }
  for (int e = tid; e < 16 * 256; e += 256) {
    int m = e >> 8, n = e & 255;
    sH[0][m][n] = (_Float16)p.hu0[(size_t)(rbase + m) * 256 + n];
    sH[1][m][n] = (_Float16)p.hl0[(size_t)(rbase + m) * 256 + n];
  }
  __syncthreads();

  const size_t w2off[4] = {OFF_W2RA, OFF_W2LA, OFF_W2RL, OFF_W2LL};

  for (int t = 0; t < T; ++t) {
    // ---- phase 1: build padded GRU inputs xu (34->64), xl (46->64) ----
    for (int e = tid; e < 16 * 64; e += 256) {
      int m = e >> 6, c = e & 63;
      float xu = 0.f, xl = 0.f;
      if (c < 31)      xu = sX[m][c];
      else if (c < 34) xu = sX[m][61 + (c - 31)];
      if (c < 13)      xl = sX[m][c];
      else if (c < 46) xl = sX[m][31 + (c - 13)];
      sXU[m][c] = (_Float16)xu;
      sXL[m][c] = (_Float16)xl;
    }
    __syncthreads();

    // ---- phase 2: GRUs (upper then lower) ----
    for (int g = 0; g < 2; ++g) {
      const size_t offIh  = g ? OFF_LO_WIH : OFF_UP_WIH;
      const size_t offHh  = g ? OFF_LO_WHH : OFF_UP_WHH;
      const float* bih    = g ? p.bih_l : p.bih_u;
      const float* bhh    = g ? p.bhh_l : p.bhh_u;
      const _Float16* xin = g ? &sXL[0][0] : &sXU[0][0];
      const _Float16* hg  = &sH[g][0][0];

      for (int nt = wv; nt < 48; nt += 8) {           // 768 gate cols = 48 N-tiles
        const int nloc = lane & 15;
        const int n = nt * 16 + nloc;
        v8f ci = bias_init(bih[n]);                   // input-path accumulator
        v8f ch = bias_init(bhh[n]);                   // hidden-path accumulator
        const _Float16* wih = p.w + offIh + (size_t)nt * 2 * FR;
#pragma unroll
        for (int kt = 0; kt < 2; ++kt) {
          v16h a = wmma_load_a(xin, 64, kt, lane);
          v16h b = wmma_load_b(wih + (size_t)kt * FR, lane);
          ci = WMMA_F16(a, b, ci);
        }
        const _Float16* whh = p.w + offHh + (size_t)nt * 8 * FR;
#pragma unroll
        for (int kt = 0; kt < 8; ++kt) {
          v16h a = wmma_load_a(hg, 256, kt, lane);
          v16h b = wmma_load_b(whh + (size_t)kt * FR, lane);
          ch = WMMA_F16(a, b, ch);
        }
        const int mb = (lane & 16) ? 8 : 0;
        if (nt < 32) {                                 // r,z gates: i+h combined
          const int col = nt * 16 + nloc;
#pragma unroll
          for (int v = 0; v < 8; ++v) sB[mb + v][col] = (_Float16)(ci[v] + ch[v]);
        } else {                                       // n gate: keep i and h separate
          const int col = (nt - 32) * 16 + nloc;
#pragma unroll
          for (int v = 0; v < 8; ++v) { sA[mb + v][col] = ci[v]; sA[mb + v][256 + col] = ch[v]; }
        }
      }
      __syncthreads();

      for (int e = tid; e < 16 * 256; e += 256) {      // GRU elementwise + h update
        int m = e >> 8, n2 = e & 255;
        float r  = sigmoidf_((float)sB[m][n2]);
        float z  = sigmoidf_((float)sB[m][256 + n2]);
        float nn = tanhf(sA[m][n2] + r * sA[m][256 + n2]);
        float hp = (float)sH[g][m][n2];
        sH[g][m][n2] = (_Float16)((1.f - z) * nn + z * hp);
      }
      __syncthreads();
    }

    // ---- phase 3: level-3 linears ub = hu@W3a.T, lb = hl@W3l.T -> sA[.][0..511] ----
    for (int nt = wv; nt < 32; nt += 8) {
      const int isU = (nt < 16);
      const int ntl = isU ? nt : nt - 16;
      const float* bias = isU ? p.b3a : p.b3l;
      const _Float16* wb = p.w + (isU ? OFF_W3A : OFF_W3L) + (size_t)ntl * 8 * FR;
      const _Float16* abase = &sH[isU ? 0 : 1][0][0];
      const int nloc = lane & 15;
      v8f c = bias_init(bias[ntl * 16 + nloc]);
#pragma unroll
      for (int kt = 0; kt < 8; ++kt) {
        v16h a = wmma_load_a(abase, 256, kt, lane);
        v16h b = wmma_load_b(wb + (size_t)kt * FR, lane);
        c = WMMA_F16(a, b, c);
      }
      const int mb = (lane & 16) ? 8 : 0;
      const int col = nt * 16 + nloc;
#pragma unroll
      for (int v = 0; v < 8; ++v) sA[mb + v][col] = c[v];
    }
    __syncthreads();
    for (int e = tid; e < 16 * 512; e += 256) {        // f32 -> f16 for next A-operand
      int m = e >> 9, c2 = e & 511;
      sB[m][c2] = (_Float16)sA[m][c2];
    }
    __syncthreads();

    // ---- phase 4: level-2 linears (ra2,la2,rl2,ll2) -> sA[.][0..255] ----
    for (int nt = wv; nt < 16; nt += 8) {
      const int g2 = nt >> 2, ntl = nt & 3;
      const float* bias = p.b2[g2];
      const _Float16* wb = p.w + w2off[g2] + (size_t)ntl * 4 * FR;
      const _Float16* abase = &sB[0][0] + g2 * 128;    // ub[:, :128] / ub[:,128:] / lb halves
      const int nloc = lane & 15;
      v8f c = bias_init(bias[ntl * 16 + nloc]);
#pragma unroll
      for (int kt = 0; kt < 4; ++kt) {
        v16h a = wmma_load_a(abase, 512, kt, lane);
        v16h b = wmma_load_b(wb + (size_t)kt * FR, lane);
        c = WMMA_F16(a, b, c);
      }
      const int mb = (lane & 16) ? 8 : 0;
      const int col = nt * 16 + nloc;
#pragma unroll
      for (int v = 0; v < 8; ++v) sA[mb + v][col] = c[v];
    }
    __syncthreads();
    for (int e = tid; e < 16 * 256; e += 256) {        // l2 f32 -> f16
      int m = e >> 8, c2 = e & 255;
      sB[m][c2] = (_Float16)sA[m][c2];
    }
    __syncthreads();

    // ---- phase 5: level-1 linears, one GEMM per wave -> sA[.][0..127] ----
    {
      const int sel[8]      = {0, 4, 1, 4, 2, 4, 3, 4};       // which W1 matrix
      const size_t w1o[5]   = {OFF_W1RA, OFF_W1LA, OFF_W1RL, OFF_W1LL, OFF_W1T};
      const int nre[5]      = {9, 9, 15, 15, 16};
      const int ocol[8]     = {0, 64, 16, 80, 32, 96, 48, 112}; // ra1,t1,la1,t2,rl1,t3,ll1,t4
      const int gm = wv;
      const int s = sel[gm];
      const float* bias = p.b1[s];
      const int nloc = lane & 15;
      float bv = (nloc < nre[s]) ? bias[nloc] : 0.f;
      v8f c = bias_init(bv);
      v16h a = wmma_load_a(&sB[0][0] + gm * 32, 512, 0, lane);
      v16h b = wmma_load_b(p.w + w1o[s], lane);
      c = WMMA_F16(a, b, c);
      const int mb = (lane & 16) ? 8 : 0;
      const int col = ocol[gm] + nloc;
#pragma unroll
      for (int v = 0; v < 8; ++v) sA[mb + v][col] = c[v];
    }
    __syncthreads();

    // ---- phase 6: torso avg, concat, residual, emit ----
    for (int e = tid; e < 16 * 64; e += 256) {
      int m = e >> 6, c = e & 63;
      float add;
      if (c < 13) {
        int j = c;
        add = 0.25f * (sA[m][64 + j] + sA[m][80 + j] + sA[m][96 + j] + sA[m][112 + j]);
      } else if (c < 22) add = sA[m][16 + (c - 13)];   // la1
      else if (c < 31)   add = sA[m][ 0 + (c - 22)];   // ra1
      else if (c < 46)   add = sA[m][48 + (c - 31)];   // ll1
      else if (c < 61)   add = sA[m][32 + (c - 46)];   // rl1
      else {
        int j = 13 + (c - 61);
        add = 0.25f * (sA[m][64 + j] + sA[m][80 + j] + sA[m][96 + j] + sA[m][112 + j]);
      }
      float xv = add + sX[m][c];
      sX[m][c] = xv;
      p.out[((size_t)(rbase + m) * T + t) * 64 + c] = xv;   // Y transposed to (B,T,64)
    }
    __syncthreads();
  }
}

// ---------------- host side ----------------
extern "C" void kernel_launch(void* const* d_in, const int* in_sizes, int n_in,
                              void* d_out, int out_size, void* d_ws, size_t ws_size,
                              hipStream_t stream) {
  _Float16* wks = (_Float16*)d_ws;

  struct PD { int idx; int N; int K; size_t off; };
  const PD descs[15] = {
      {4,  768,  34, OFF_UP_WIH}, {5,  768, 256, OFF_UP_WHH},
      {8,  768,  46, OFF_LO_WIH}, {9,  768, 256, OFF_LO_WHH},
      {12, 256, 256, OFF_W3A},    {14, 256, 256, OFF_W3L},
      {16,  64, 128, OFF_W2RA},   {18,  64, 128, OFF_W2LA},
      {20,  64, 128, OFF_W2RL},   {22,  64, 128, OFF_W2LL},
      {24,   9,  32, OFF_W1RA},   {26,   9,  32, OFF_W1LA},
      {28,  15,  32, OFF_W1RL},   {30,  15,  32, OFF_W1LL},
      {32,  16,  32, OFF_W1T},
  };
  for (int i = 0; i < 15; ++i) {
    const PD& d = descs[i];
    int NT = (d.N + 15) / 16, KT = (d.K + 31) / 32;
    int total = NT * KT * 512;
    pack_w_kernel<<<(total + 255) / 256, 256, 0, stream>>>(
        (const float*)d_in[d.idx], wks + d.off, d.N, d.K, NT, KT);
  }

  KParams p;
  p.hu0    = (const float*)d_in[0];
  p.hl0    = (const float*)d_in[1];
  p.gt     = (const float*)d_in[2];
  p.tsteps = (const int*)d_in[3];
  p.w      = wks;
  p.bih_u  = (const float*)d_in[6];
  p.bhh_u  = (const float*)d_in[7];
  p.bih_l  = (const float*)d_in[10];
  p.bhh_l  = (const float*)d_in[11];
  p.b3a    = (const float*)d_in[13];
  p.b3l    = (const float*)d_in[15];
  p.b2[0]  = (const float*)d_in[17];  // b2ra
  p.b2[1]  = (const float*)d_in[19];  // b2la
  p.b2[2]  = (const float*)d_in[21];  // b2rl
  p.b2[3]  = (const float*)d_in[23];  // b2ll
  p.b1[0]  = (const float*)d_in[25];  // b1ra
  p.b1[1]  = (const float*)d_in[27];  // b1la
  p.b1[2]  = (const float*)d_in[29];  // b1rl
  p.b1[3]  = (const float*)d_in[31];  // b1ll
  p.b1[4]  = (const float*)d_in[33];  // b1t
  p.out    = (float*)d_out;

  const int B = in_sizes[0] / 256;    // h_upper is (B, 256)
  const int blocks = B / 16;
  veldec_main<<<blocks, 256, 0, stream>>>(p);
}
